// PerformerAttention_19464791785690
// MI455X (gfx1250) — matrix operand order repaired and checked by reading, not compiled, not run on hardware
//
#include <hip/hip_runtime.h>
#include <hip/hip_bf16.h>
#include <math.h>

// ---------------------------------------------------------------------------
// CDNA5 (gfx1250) Performer attention, wave32 + v_wmma_f32_16x16x32_f16.
// Block = 256 threads = 8 wave32s arranged 4 (M) x 2 (N).
// Block tile: 128 x BN (BN = 128 or 64), K-tile = 32, double-buffered LDS.
// A staged in LDS row-major [row][k]; B staged TRANSPOSED [n][k] so both
// fragment loads are contiguous 16B ds_load_b128s in the contiguous k = 16*(lane/16) + s order used for A and B.
// Staging uses 128-bit global loads (b128) wherever the coalesced axis allows.
// ---------------------------------------------------------------------------

typedef __attribute__((ext_vector_type(16))) _Float16 v16h;
typedef __attribute__((ext_vector_type(8)))  _Float16 v8h;
typedef __attribute__((ext_vector_type(4)))  _Float16 v4h;
typedef __attribute__((ext_vector_type(8)))  float    v8f;

#define TB  256     // threads per block
#define BK  32      // K tile (matches 16x16x32 f16 WMMA)
#define LDH 40      // LDS row stride in halves (32 + 8 pad, keeps 16B align)

__device__ __forceinline__ v8f wmma16(v16h a, v16h b, v8f c) {
  // D = A(16x32 f16) x B(32x16 f16) + C(16x16 f32)
  return __builtin_amdgcn_wmma_f32_16x16x32_f16(false, a, false, b,
                                                (short)0, c, false, false);
}

// A fragment: 16x32 f16.  lanes 0-15: row = lane,  K = 0..15
//             lanes 16-31: row = lane-16, K = 16..31
__device__ __forceinline__ v16h load_a_frag(const _Float16* As, int row0, int lane) {
  const int r  = row0 + (lane & 15);
  const int kb = (lane & 16) ? 16 : 0;
  const _Float16* p = As + r * LDH + kb;
  v16h a;
#pragma unroll
  for (int i = 0; i < 8; ++i) { a[i] = p[i]; a[i + 8] = p[i + 8]; }
  return a;
}

// B fragment: 32x16 f16 from transposed LDS tile Bts[n][k].
// lanes 0-15: col = lane, K = 0..15; lanes 16-31: col = lane-16, K = 16..31
__device__ __forceinline__ v16h load_b_frag(const _Float16* Bts, int col0, int lane) {
  const int n  = col0 + (lane & 15);
  const int kb = (lane & 16) ? 16 : 0;
  const _Float16* p = Bts + n * LDH + kb;
  v16h b;
#pragma unroll
  for (int i = 0; i < 16; ++i) b[i] = p[i];
  return b;
}

template <int NT>
__device__ __forceinline__ void zero_acc(v8f (*acc)[NT]) {
#pragma unroll
  for (int m = 0; m < 2; ++m)
#pragma unroll
    for (int n = 0; n < NT; ++n)
#pragma unroll
      for (int e = 0; e < 8; ++e) acc[m][n][e] = 0.0f;
}

// One K-step for a wave: 2 M-tiles x NT N-tiles of 16x16.
template <int NT>
__device__ __forceinline__ void mma_step(const _Float16* As, const _Float16* Bts,
                                         int wrow, int wcol, int lane,
                                         v8f (*acc)[NT]) {
  v16h a0 = load_a_frag(As, wrow * 32,      lane);
  v16h a1 = load_a_frag(As, wrow * 32 + 16, lane);
#pragma unroll
  for (int nt = 0; nt < NT; ++nt) {
    v16h b = load_b_frag(Bts, wcol * (NT * 16) + nt * 16, lane);
    acc[0][nt] = wmma16(a0, b, acc[0][nt]);
    acc[1][nt] = wmma16(a1, b, acc[1][nt]);
  }
}

// ---- LDS staging helpers (all dims divide evenly; no bounds checks) -------

// A row-major f32 -> LDS f16 [r][k]: float4 loads, packed v4h (b64) LDS stores.
__device__ __forceinline__ void stage_a_f32(const float* A, int strideA, int rows,
                                            _Float16* As, int tid) {
  for (int idx = tid; idx < rows * 8; idx += TB) {
    const int r = idx >> 3, k4 = (idx & 7) << 2;
    const float4 f = *(const float4*)(A + (size_t)r * strideA + k4);
    v4h h;
    h[0] = (_Float16)f.x; h[1] = (_Float16)f.y;
    h[2] = (_Float16)f.z; h[3] = (_Float16)f.w;
    *(v4h*)(As + r * LDH + k4) = h;
  }
}

// A row-major f16 -> LDS [r][k]: 16B copies (global b128 -> ds b128).
__device__ __forceinline__ void stage_a_f16(const _Float16* A, int strideA, int rows,
                                            _Float16* As, int tid) {
  for (int idx = tid; idx < rows * 4; idx += TB) {
    const int r = idx >> 2, k8 = (idx & 3) << 3;
    *(v8h*)(As + r * LDH + k8) = *(const v8h*)(A + (size_t)r * strideA + k8);
  }
}

// A stored [k][r] in memory -> LDS [r][k] (transpose): 8B loads along r.
__device__ __forceinline__ void stage_at_f16(const _Float16* A, int strideA, int rows,
                                             _Float16* As, int tid) {
  const int rq = rows >> 2;
  for (int idx = tid; idx < rq * BK; idx += TB) {
    const int r4 = (idx % rq) << 2, k = idx / rq;
    const v4h h = *(const v4h*)(A + (size_t)k * strideA + r4);
#pragma unroll
    for (int j = 0; j < 4; ++j) As[(r4 + j) * LDH + k] = h[j];
  }
}

// B stored [k][n] f32 -> LDS transposed [n][k] f16: float4 loads along n.
__device__ __forceinline__ void stage_bt_f32(const float* B, int strideB, int cols,
                                             _Float16* Bts, int tid) {
  const int nq = cols >> 2;
  for (int idx = tid; idx < nq * BK; idx += TB) {
    const int n4 = (idx % nq) << 2, k = idx / nq;
    const float4 f = *(const float4*)(B + (size_t)k * strideB + n4);
    Bts[(n4 + 0) * LDH + k] = (_Float16)f.x;
    Bts[(n4 + 1) * LDH + k] = (_Float16)f.y;
    Bts[(n4 + 2) * LDH + k] = (_Float16)f.z;
    Bts[(n4 + 3) * LDH + k] = (_Float16)f.w;
  }
}

// B stored [k][n] f16 -> LDS transposed [n][k]: 8B loads along n.
__device__ __forceinline__ void stage_bt_f16(const _Float16* B, int strideB, int cols,
                                             _Float16* Bts, int tid) {
  const int nq = cols >> 2;
  for (int idx = tid; idx < nq * BK; idx += TB) {
    const int n4 = (idx % nq) << 2, k = idx / nq;
    const v4h h = *(const v4h*)(B + (size_t)k * strideB + n4);
#pragma unroll
    for (int j = 0; j < 4; ++j) Bts[(n4 + j) * LDH + k] = h[j];
  }
}

// ---------------------------------------------------------------------------
// Problem constants: B=4, S=4096, D=1024, H=16, hd=64, M=256
// ---------------------------------------------------------------------------

// K1: Q/K/V = x @ W{q,k,v} + b, written f16 in (z, b, h, s, hd) layout.
__global__ __launch_bounds__(TB)
void qkv_gemm(const float* __restrict__ x,
              const float* __restrict__ Wq, const float* __restrict__ bq,
              const float* __restrict__ Wk, const float* __restrict__ bk,
              const float* __restrict__ Wv, const float* __restrict__ bv,
              _Float16* __restrict__ qkv) {
  __shared__ _Float16 As[2][128 * LDH];
  __shared__ _Float16 Bts[2][128 * LDH];
  const int tid = threadIdx.x, lane = tid & 31, wid = tid >> 5;
  const int wrow = wid & 3, wcol = wid >> 2;
  const int z = blockIdx.z;
  const float* W    = (z == 0) ? Wq : (z == 1) ? Wk : Wv;
  const float* bias = (z == 0) ? bq : (z == 1) ? bk : bv;
  const int row0 = blockIdx.x * 128;
  const int n0   = blockIdx.y * 128;

  v8f acc[2][4];
  zero_acc<4>(acc);

  stage_a_f32(x + (size_t)row0 * 1024, 1024, 128, As[0], tid);
  stage_bt_f32(W + n0, 1024, 128, Bts[0], tid);
  __syncthreads();

  for (int kt = 0; kt < 1024; kt += BK) {
    const int cur = (kt >> 5) & 1;
    if (kt + BK < 1024) {   // stage next tile while WMMAs consume current one
      stage_a_f32(x + (size_t)row0 * 1024 + kt + BK, 1024, 128, As[cur ^ 1], tid);
      stage_bt_f32(W + (size_t)(kt + BK) * 1024 + n0, 1024, 128, Bts[cur ^ 1], tid);
      if (kt + 2 * BK < 1024) {  // gfx1250 global_prefetch_b8 of the tile after
        __builtin_prefetch(x + (size_t)(row0 + (tid & 127)) * 1024 + kt + 2 * BK, 0, 1);
        __builtin_prefetch(W + (size_t)(kt + 2 * BK + (tid >> 3)) * 1024 + n0, 0, 1);
      }
    }
    mma_step<4>(As[cur], Bts[cur], wrow, wcol, lane, acc);
    __syncthreads();
  }

#pragma unroll
  for (int mt = 0; mt < 2; ++mt)
#pragma unroll
    for (int nt = 0; nt < 4; ++nt)
#pragma unroll
      for (int i = 0; i < 8; ++i) {
        const int m = row0 + wrow * 32 + mt * 16 + i + ((lane >> 4) << 3);
        const int n = n0 + wcol * 64 + nt * 16 + (lane & 15);
        const float v = acc[mt][nt][i] + bias[n];
        const int b = m >> 12, s = m & 4095;
        const int h = n >> 6,  e = n & 63;
        qkv[((((size_t)z * 4 + b) * 16 + h) * 4096 + s) * 64 + e] = (_Float16)v;
      }
}

// K2: q'/k' = (elu(Qh @ R) + 1) / sqrt(M); per-head (4096x64)@(64x256).
__global__ __launch_bounds__(TB)
void feature_gemm(const _Float16* __restrict__ qkv,
                  const float* __restrict__ R,
                  _Float16* __restrict__ prime) {
  __shared__ _Float16 As[2][128 * LDH];
  __shared__ _Float16 Bts[2][128 * LDH];
  const int tid = threadIdx.x, lane = tid & 31, wid = tid >> 5;
  const int wrow = wid & 3, wcol = wid >> 2;
  const int zz = blockIdx.z >> 6;     // 0 = q, 1 = k
  const int bh = blockIdx.z & 63;
  const int row0 = blockIdx.x * 128;  // s
  const int n0   = blockIdx.y * 128;  // m
  const _Float16* Abase = qkv + (((size_t)zz * 64 + bh) * 4096 + row0) * 64;

  v8f acc[2][4];
  zero_acc<4>(acc);

  stage_a_f16(Abase, 64, 128, As[0], tid);
  stage_bt_f32(R + n0, 256, 128, Bts[0], tid);
  __syncthreads();

  for (int kt = 0; kt < 64; kt += BK) {
    const int cur = (kt >> 5) & 1;
    if (kt + BK < 64) {
      stage_a_f16(Abase + kt + BK, 64, 128, As[cur ^ 1], tid);
      stage_bt_f32(R + (size_t)(kt + BK) * 256 + n0, 256, 128, Bts[cur ^ 1], tid);
    }
    mma_step<4>(As[cur], Bts[cur], wrow, wcol, lane, acc);
    __syncthreads();
  }

  _Float16* Obase = prime + ((size_t)zz * 64 + bh) * 4096 * 256;
#pragma unroll
  for (int mt = 0; mt < 2; ++mt)
#pragma unroll
    for (int nt = 0; nt < 4; ++nt)
#pragma unroll
      for (int i = 0; i < 8; ++i) {
        const int s = row0 + wrow * 32 + mt * 16 + i + ((lane >> 4) << 3);
        const int n = n0 + wcol * 64 + nt * 16 + (lane & 15);
        float v = acc[mt][nt][i];
        v = (v > 0.0f) ? (v + 1.0f) : __expf(v);  // elu(v)+1
        v *= 0.0625f;                              // 1/sqrt(256)
        Obase[(size_t)s * 256 + n] = (_Float16)v;
      }
}

// K3: k_sum[bh][m] = sum_s k'[bh][s][m]
__global__ __launch_bounds__(TB)
void ksum_kernel(const _Float16* __restrict__ kprime, float* __restrict__ ksum) {
  const int bh = blockIdx.x, m = threadIdx.x;
  const _Float16* p = kprime + (size_t)bh * 4096 * 256 + m;
  float s = 0.0f;
#pragma unroll 8
  for (int i = 0; i < 4096; ++i) s += (float)p[(size_t)i * 256];
  ksum[bh * 256 + m] = s;
}

// K4: kv[bh] = k'[bh]^T @ v[bh], (256x4096)@(4096x64), split-K=8 with f32 atomics.
__global__ __launch_bounds__(TB)
void kv_gemm(const _Float16* __restrict__ kprime,
             const _Float16* __restrict__ vhead,
             float* __restrict__ kv) {
  __shared__ _Float16 As[2][128 * LDH];
  __shared__ _Float16 Bts[2][64 * LDH];
  const int tid = threadIdx.x, lane = tid & 31, wid = tid >> 5;
  const int wrow = wid & 3, wcol = wid >> 2;
  const int bh = blockIdx.z;
  const int m0 = blockIdx.x * 128;
  const int s0base = blockIdx.y * 512;
  const _Float16* K = kprime + (size_t)bh * 4096 * 256;
  const _Float16* V = vhead  + (size_t)bh * 4096 * 64;

  v8f acc[2][2];
  zero_acc<2>(acc);

  stage_at_f16(K + (size_t)s0base * 256 + m0, 256, 128, As[0], tid);
  stage_bt_f16(V + (size_t)s0base * 64, 64, 64, Bts[0], tid);
  __syncthreads();

  for (int kt = 0; kt < 512; kt += BK) {
    const int cur = (kt >> 5) & 1;
    if (kt + BK < 512) {
      const int s1 = s0base + kt + BK;
      stage_at_f16(K + (size_t)s1 * 256 + m0, 256, 128, As[cur ^ 1], tid);
      stage_bt_f16(V + (size_t)s1 * 64, 64, 64, Bts[cur ^ 1], tid);
    }
    mma_step<2>(As[cur], Bts[cur], wrow, wcol, lane, acc);
    __syncthreads();
  }

#pragma unroll
  for (int mt = 0; mt < 2; ++mt)
#pragma unroll
    for (int nt = 0; nt < 2; ++nt)
#pragma unroll
      for (int i = 0; i < 8; ++i) {
        const int m = m0 + wrow * 32 + mt * 16 + i + ((lane >> 4) << 3);
        const int n = wcol * 32 + nt * 16 + (lane & 15);
        atomicAdd(&kv[((size_t)bh * 256 + m) * 64 + n], acc[mt][nt][i]);
      }
}

// K5: normalizer[bh][s] = max(sum_m q'[s][m] * k_sum[m], 1e-6); one wave per row.
__global__ __launch_bounds__(TB)
void norm_kernel(const _Float16* __restrict__ qprime,
                 const float* __restrict__ ksum,
                 float* __restrict__ normb) {
  const int wid = threadIdx.x >> 5, lane = threadIdx.x & 31;
  const int g = blockIdx.x * 8 + wid;      // 0 .. 64*4096-1
  const int bh = g >> 12;
  const _Float16* q = qprime + (size_t)g * 256;
  const float* ks = ksum + bh * 256;
  float acc = 0.0f;
#pragma unroll
  for (int j = 0; j < 8; ++j) {
    const int m = lane + j * 32;
    acc += (float)q[m] * ks[m];
  }
#pragma unroll
  for (int off = 16; off > 0; off >>= 1) acc += __shfl_xor(acc, off, 32);
  if (lane == 0) normb[g] = fmaxf(acc, 1e-6f);
}

// K6: att = (q' @ kv) / normalizer, written f16 in (b, s, h, hd) layout.
__global__ __launch_bounds__(TB)
void outhead_gemm(const _Float16* __restrict__ qprime,
                  const float* __restrict__ kv,
                  const float* __restrict__ normb,
                  _Float16* __restrict__ att) {
  __shared__ _Float16 As[2][128 * LDH];
  __shared__ _Float16 Bts[2][64 * LDH];
  const int tid = threadIdx.x, lane = tid & 31, wid = tid >> 5;
  const int wrow = wid & 3, wcol = wid >> 2;
  const int bh = blockIdx.z;
  const int b = bh >> 4, h = bh & 15;
  const int row0 = blockIdx.x * 128;
  const _Float16* Q  = qprime + ((size_t)bh * 4096 + row0) * 256;
  const float*    KV = kv + (size_t)bh * 256 * 64;

  v8f acc[2][2];
  zero_acc<2>(acc);

  stage_a_f16(Q, 256, 128, As[0], tid);
  stage_bt_f32(KV, 64, 64, Bts[0], tid);
  __syncthreads();

  for (int kt = 0; kt < 256; kt += BK) {
    const int cur = (kt >> 5) & 1;
    if (kt + BK < 256) {
      stage_a_f16(Q + kt + BK, 256, 128, As[cur ^ 1], tid);
      stage_bt_f32(KV + (size_t)(kt + BK) * 64, 64, 64, Bts[cur ^ 1], tid);
    }
    mma_step<2>(As[cur], Bts[cur], wrow, wcol, lane, acc);
    __syncthreads();
  }

#pragma unroll
  for (int mt = 0; mt < 2; ++mt)
#pragma unroll
    for (int nt = 0; nt < 2; ++nt)
#pragma unroll
      for (int i = 0; i < 8; ++i) {
        const int s = row0 + wrow * 32 + mt * 16 + i + ((lane >> 4) << 3);
        const int n = wcol * 32 + nt * 16 + (lane & 15);
        const float inv = 1.0f / normb[(size_t)bh * 4096 + s];
        att[(((size_t)b * 4096 + s) * 16 + h) * 64 + n] = (_Float16)(acc[mt][nt][i] * inv);
      }
}

// K7: out = att @ Wo + bo (f32 result).
__global__ __launch_bounds__(TB)
void final_gemm(const _Float16* __restrict__ att,
                const float* __restrict__ Wo, const float* __restrict__ bo,
                float* __restrict__ out) {
  __shared__ _Float16 As[2][128 * LDH];
  __shared__ _Float16 Bts[2][128 * LDH];
  const int tid = threadIdx.x, lane = tid & 31, wid = tid >> 5;
  const int wrow = wid & 3, wcol = wid >> 2;
  const int row0 = blockIdx.x * 128;
  const int n0   = blockIdx.y * 128;

  v8f acc[2][4];
  zero_acc<4>(acc);

  stage_a_f16(att + (size_t)row0 * 1024, 1024, 128, As[0], tid);
  stage_bt_f32(Wo + n0, 1024, 128, Bts[0], tid);
  __syncthreads();

  for (int kt = 0; kt < 1024; kt += BK) {
    const int cur = (kt >> 5) & 1;
    if (kt + BK < 1024) {
      stage_a_f16(att + (size_t)row0 * 1024 + kt + BK, 1024, 128, As[cur ^ 1], tid);
      stage_bt_f32(Wo + (size_t)(kt + BK) * 1024 + n0, 1024, 128, Bts[cur ^ 1], tid);
      if (kt + 2 * BK < 1024) {
        __builtin_prefetch(att + (size_t)(row0 + (tid & 127)) * 1024 + kt + 2 * BK, 0, 1);
        __builtin_prefetch(Wo + (size_t)(kt + 2 * BK + (tid >> 3)) * 1024 + n0, 0, 1);
      }
    }
    mma_step<4>(As[cur], Bts[cur], wrow, wcol, lane, acc);
    __syncthreads();
  }

#pragma unroll
  for (int mt = 0; mt < 2; ++mt)
#pragma unroll
    for (int nt = 0; nt < 4; ++nt)
#pragma unroll
      for (int i = 0; i < 8; ++i) {
        const int m = row0 + wrow * 32 + mt * 16 + i + ((lane >> 4) << 3);
        const int n = n0 + wcol * 64 + nt * 16 + (lane & 15);
        out[(size_t)m * 1024 + n] = acc[mt][nt][i] + bo[n];
      }
}

// ---------------------------------------------------------------------------

extern "C" void kernel_launch(void* const* d_in, const int* in_sizes, int n_in,
                              void* d_out, int out_size, void* d_ws, size_t ws_size,
                              hipStream_t stream) {
  (void)in_sizes; (void)n_in; (void)out_size; (void)ws_size;
  const float* x  = (const float*)d_in[0];
  const float* Wq = (const float*)d_in[1];
  const float* bq = (const float*)d_in[2];
  const float* Wk = (const float*)d_in[3];
  const float* bk = (const float*)d_in[4];
  const float* Wv = (const float*)d_in[5];
  const float* bv = (const float*)d_in[6];
  const float* Wo = (const float*)d_in[7];
  const float* bo = (const float*)d_in[8];
  const float* R  = (const float*)d_in[9];
  float* out = (float*)d_out;

  // Workspace layout
  char* ws = (char*)d_ws;
  _Float16* qkv   = (_Float16*)ws; ws += (size_t)3 * 64 * 4096 * 64 * sizeof(_Float16);   // 100.7 MB
  _Float16* prime = (_Float16*)ws; ws += (size_t)2 * 64 * 4096 * 256 * sizeof(_Float16);  // 268.4 MB
  float*    kv    = (float*)ws;    ws += (size_t)64 * 256 * 64 * sizeof(float);           // 4 MB
  float*    ksum  = (float*)ws;    ws += (size_t)64 * 256 * sizeof(float);                // 64 KB
  float*    normb = (float*)ws;    ws += (size_t)64 * 4096 * sizeof(float);               // 1 MB
  _Float16* att   = (_Float16*)ws; ws += (size_t)16384 * 1024 * sizeof(_Float16);         // 33.6 MB

  _Float16* kprime = prime + (size_t)64 * 4096 * 256;
  _Float16* vhead  = qkv + (size_t)2 * 64 * 4096 * 64;

  const dim3 blk(TB);
  qkv_gemm    <<<dim3(128, 8, 3),  blk, 0, stream>>>(x, Wq, bq, Wk, bk, Wv, bv, qkv);
  feature_gemm<<<dim3(32, 2, 128), blk, 0, stream>>>(qkv, R, prime);
  ksum_kernel <<<dim3(64),         blk, 0, stream>>>(kprime, ksum);
  hipMemsetAsync(kv, 0, (size_t)64 * 256 * 64 * sizeof(float), stream);
  kv_gemm     <<<dim3(2, 8, 64),   blk, 0, stream>>>(kprime, vhead, kv);
  norm_kernel <<<dim3(32768),      blk, 0, stream>>>(prime, ksum, normb);
  outhead_gemm<<<dim3(32, 1, 64),  blk, 0, stream>>>(prime, kv, normb, att);
  final_gemm  <<<dim3(128, 8),     blk, 0, stream>>>(att, Wo, bo, out);
}
